// CrossAttention_16260746183230
// MI455X (gfx1250) — compile-verified, matrix operand order checked by reading, not run on hardware
//
#include <hip/hip_runtime.h>
#include <hip/hip_bf16.h>

// Set to 0 to fall back to the manual cooperative LDS staging loop if the
// tensor_load_to_lds inline-asm syntax is rejected by this toolchain.
#define USE_TDM 1

#define DEVI __device__ __forceinline__

typedef __attribute__((ext_vector_type(16))) __bf16 v16bf;
typedef __attribute__((ext_vector_type(8)))  __bf16 v8bf;
typedef __attribute__((ext_vector_type(8)))  float  v8f;
typedef uint32_t u32x4 __attribute__((ext_vector_type(4)));
typedef uint32_t u32x8 __attribute__((ext_vector_type(8)));

constexpr int Bc = 8;
constexpr int Lc = 2048;
constexpr int Ec = 128;

// ---------------------------------------------------------------------------
// WMMA 16x16x32 bf16 A/B tile loader.
// A-layout (16-bit, 16x32): lane l -> row m = l%16, half-group g = l/16.
// Half-slot s (0..15) holds K = 16*(s/8) + 8*g + (s%8)  ==>  per lane the
// tile is two contiguous 8-halfword runs at [8g .. 8g+7] and [16+8g .. 16+8g+7].
// B-layout is symmetric with lane = N, so X @ Y^T needs only ROWS of X and Y.
// ---------------------------------------------------------------------------
DEVI v16bf load_ab(const __bf16* p, int g) {
  v8bf lo = *(const v8bf*)(p + 8 * g);
  v8bf hi = *(const v8bf*)(p + 16 + 8 * g);
  return __builtin_shufflevector(lo, hi, 0, 1, 2, 3, 4, 5, 6, 7,
                                 8, 9, 10, 11, 12, 13, 14, 15);
}

DEVI v8f wmma_bf16(v16bf a, v16bf b, v8f c) {
  return __builtin_amdgcn_wmma_f32_16x16x32_bf16(false, a, false, b,
                                                 (short)0, c, false, false);
}

DEVI v8f vzero8() {
  v8f z = {0.f, 0.f, 0.f, 0.f, 0.f, 0.f, 0.f, 0.f};
  return z;
}

#if USE_TDM
// ---------------------------------------------------------------------------
// TDM: DMA a 2D tile (rows x row_elems of 2-byte elements, row stride in
// elements) from global memory into LDS. Builds the D# descriptor (ISA ch.8):
//   group0: count=1 | lds_addr | global_addr[56:0] | type=2
//   group1: data_size=1(2B), tensor_dim0/1, tile_dim0/1, dim0_stride; 2D so
//           VADDR2/VADDR3 are omitted (NULL).
// Low 32 bits of a flat pointer to __shared__ memory are the LDS byte address
// (ISA 10.2).
// ---------------------------------------------------------------------------
DEVI void tdm_load_2d_bf16(const void* gptr, uint32_t lds_addr,
                           uint32_t rows, uint32_t row_elems,
                           uint32_t row_stride_elems) {
  const uint64_t ga = (uint64_t)(uintptr_t)gptr;
  u32x4 g0;
  g0[0] = 1u;                                   // count=1 (valid user D#)
  g0[1] = lds_addr;                             // LDS destination (bytes)
  g0[2] = (uint32_t)ga;                         // global_addr[31:0]
  g0[3] = (uint32_t)((ga >> 32) & 0x01FFFFFFu)  // global_addr[56:32]
          | (2u << 30);                         // type=2 ("image")
  u32x8 g1;
  g1[0] = 1u << 16;                             // data_size=1 -> 2 bytes
  g1[1] = (row_elems & 0xFFFFu) << 16;          // tensor_dim0[15:0] @ bits63:48
  g1[2] = (row_elems >> 16)                     // tensor_dim0[31:16]
          | ((rows & 0xFFFFu) << 16);           // tensor_dim1[15:0]
  g1[3] = (rows >> 16)                          // tensor_dim1[31:16]
          | ((row_elems & 0xFFFFu) << 16);      // tile_dim0
  g1[4] = rows & 0xFFFFu;                       // tile_dim1 (tile_dim2=0)
  g1[5] = row_stride_elems;                     // tensor_dim0_stride[31:0]
  g1[6] = 0u;                                   // stride hi, dim1_stride lo
  g1[7] = 0u;
  asm volatile("tensor_load_to_lds %0, %1"
               :: "s"(g0), "s"(g1)
               : "memory");
}
#endif

// ---------------------------------------------------------------------------
// Kernel 1: bf16 staging buffers.
// ---------------------------------------------------------------------------
__global__ void __launch_bounds__(256)
k_prep(const float* __restrict__ Uq, const float* __restrict__ Uid,
       const float* __restrict__ Wcw,
       __bf16* __restrict__ Uqb, __bf16* __restrict__ Uwb,
       __bf16* __restrict__ UqT, __bf16* __restrict__ UidT) {
  const int idx = blockIdx.x * blockDim.x + threadIdx.x;  // < B*L*E
  const int e  = idx % Ec;
  const int bi = idx / Ec;
  const int i  = bi % Lc;
  const int b  = bi / Lc;
  const float uq  = Uq[idx];
  const float uid = Uid[idx];
  Uqb[idx] = (__bf16)uq;
  Uwb[idx] = (__bf16)(uid * Wcw[2 * Ec + e]);
  const size_t t = ((size_t)b * Ec + e) * Lc + i;
  UqT[t]  = (__bf16)uq;
  UidT[t] = (__bf16)uid;
}

// ---------------------------------------------------------------------------
// Kernel 2: s_id[b,i] = Uid[b,i,:]·w_id ; s_q[b,j] = Uq[b,j,:]·w_q
// ---------------------------------------------------------------------------
__global__ void __launch_bounds__(256)
k_rowdots(const float* __restrict__ Uq, const float* __restrict__ Uid,
          const float* __restrict__ Wcw,
          float* __restrict__ sid, float* __restrict__ sq) {
  const int row  = blockIdx.x * 8 + (threadIdx.x >> 5);  // B*L rows
  const int lane = threadIdx.x & 31;
  const float4 a  = ((const float4*)(Uid + (size_t)row * Ec))[lane];
  const float4 b  = ((const float4*)(Uq  + (size_t)row * Ec))[lane];
  const float4 wa = ((const float4*)(Wcw))[lane];
  const float4 wb = ((const float4*)(Wcw + Ec))[lane];
  float d1 = a.x * wa.x + a.y * wa.y + a.z * wa.z + a.w * wa.w;
  float d2 = b.x * wb.x + b.y * wb.y + b.z * wb.z + b.w * wb.w;
#pragma unroll
  for (int off = 16; off > 0; off >>= 1) {
    d1 += __shfl_xor(d1, off, 32);
    d2 += __shfl_xor(d2, off, 32);
  }
  if (lane == 0) { sid[row] = d1; sq[row] = d2; }
}

// ---------------------------------------------------------------------------
// Kernel 3: scores + softmax over axis i (per column j) -> P (bf16).
// ---------------------------------------------------------------------------
__global__ void __launch_bounds__(32)
k_scores_softmax(const __bf16* __restrict__ Uwb, const __bf16* __restrict__ Uqb,
                 const float* __restrict__ sid, const float* __restrict__ sq,
                 const float* __restrict__ mask, const float* __restrict__ Wcb,
                 __bf16* __restrict__ P) {
  const int wg = blockIdx.x;            // B * (L/16)
  const int b  = wg / (Lc / 16);
  const int j0 = (wg % (Lc / 16)) * 16;
  const int l  = threadIdx.x;
  const int lm = l & 15, g = l >> 4;
  const int j  = j0 + lm;
  const float bias = Wcb[0];
  const __bf16* Uwb_b = Uwb + (size_t)b * Lc * Ec;
  const float*  sid_b = sid + b * Lc;
  const float   sqj   = sq[b * Lc + j];
  __bf16* P_b = P + (size_t)b * Lc * Lc;

  v16bf Bk[4];
  {
    const __bf16* brow = Uqb + ((size_t)b * Lc + j) * Ec;
#pragma unroll
    for (int ks = 0; ks < 4; ++ks) Bk[ks] = load_ab(brow + 32 * ks, g);
  }

  float m = -3.0e38f, ssum = 0.f;
  for (int it = 0; it < Lc / 16; ++it) {
    const int i0 = it * 16;
    v8f acc = vzero8();
    const __bf16* arow = Uwb_b + (size_t)(i0 + lm) * Ec;
#pragma unroll
    for (int ks = 0; ks < 4; ++ks)
      acc = wmma_bf16(load_ab(arow + 32 * ks, g), Bk[ks], acc);
    float tmax = -3.0e38f;
#pragma unroll
    for (int r = 0; r < 8; ++r) {
      const int ir = i0 + r + 8 * g;
      const float s = (acc[r] + sid_b[ir] + sqj + bias) * mask[(size_t)ir * Lc + j];
      acc[r] = s;
      tmax = fmaxf(tmax, s);
    }
    tmax = fmaxf(tmax, __shfl_xor(tmax, 16, 32));
    const float nm = fmaxf(m, tmax);
    float p = 0.f;
#pragma unroll
    for (int r = 0; r < 8; ++r) p += __expf(acc[r] - nm);
    p += __shfl_xor(p, 16, 32);
    ssum = ssum * __expf(m - nm) + p;
    m = nm;
  }
  const float rinv = 1.f / ssum;

  for (int it = 0; it < Lc / 16; ++it) {
    const int i0 = it * 16;
    v8f acc = vzero8();
    const __bf16* arow = Uwb_b + (size_t)(i0 + lm) * Ec;
#pragma unroll
    for (int ks = 0; ks < 4; ++ks)
      acc = wmma_bf16(load_ab(arow + 32 * ks, g), Bk[ks], acc);
#pragma unroll
    for (int r = 0; r < 8; ++r) {
      const int ir = i0 + r + 8 * g;
      const float s = (acc[r] + sid_b[ir] + sqj + bias) * mask[(size_t)ir * Lc + j];
      P_b[(size_t)ir * Lc + j] = (__bf16)(__expf(s - m) * rinv);
    }
  }
}

// ---------------------------------------------------------------------------
// Kernel 4: A_D2Q = P @ Uq (K = 2048) + output chunks 0..2.
// ---------------------------------------------------------------------------
__global__ void __launch_bounds__(32)
k_d2q(const __bf16* __restrict__ P, const __bf16* __restrict__ UqT,
      const float* __restrict__ Uid, float* __restrict__ out) {
  const int wg = blockIdx.x;            // B * (L/16)
  const int b  = wg / (Lc / 16);
  const int i0 = (wg % (Lc / 16)) * 16;
  const int l = threadIdx.x, lm = l & 15, g = l >> 4;
  const __bf16* P_b   = P   + (size_t)b * Lc * Lc;
  const __bf16* UqT_b = UqT + (size_t)b * Ec * Lc;

  v8f acc[8];
#pragma unroll
  for (int t = 0; t < 8; ++t) acc[t] = vzero8();

  const __bf16* arow = P_b + (size_t)(i0 + lm) * Lc;
  for (int k0 = 0; k0 < Lc; k0 += 32) {
    __builtin_prefetch(arow + k0 + 64, 0, 0);      // global_prefetch_b8
    const v16bf a = load_ab(arow + k0, g);
#pragma unroll
    for (int t = 0; t < 8; ++t) {
      const __bf16* brow = UqT_b + (size_t)(t * 16 + lm) * Lc + k0;
      acc[t] = wmma_bf16(a, load_ab(brow, g), acc[t]);
    }
  }

  const float* Uid_b = Uid + (size_t)b * Lc * Ec;
  float* out_b = out + (size_t)b * Lc * (4 * Ec);
#pragma unroll
  for (int t = 0; t < 8; ++t) {
#pragma unroll
    for (int r = 0; r < 8; ++r) {
      const int ir = i0 + r + 8 * g;
      const int e  = t * 16 + lm;
      const float v = acc[t][r];
      const float u = Uid_b[(size_t)ir * Ec + e];
      float* o = out_b + (size_t)ir * (4 * Ec);
      o[e]          = u;
      o[Ec + e]     = v;
      o[2 * Ec + e] = u * v;
    }
  }
}

// ---------------------------------------------------------------------------
// Kernel 5 (dominant, ~137 of 163 GFLOP): fused M = P @ P^T, A_Q2D = M @ Uid.
// WG = 4 waves / 64 i-rows. Per 64-wide k-block:
//   1) TDM (tensor_load_to_lds) stages P[k0..k0+63][0..2047] (256 KB) into
//      LDS — one DMA descriptor issued by wave 0 replaces 16384 load/store
//      pairs; waited with s_wait_tensorcnt then a workgroup barrier.
//   2) each wave: M strip (16 i x 64 k) = P_i @ Pk^T, B tiles via ds_load_b128
//   3) transpose M strip C-layout->A-layout via per-wave 2 KB LDS (+bf16 cvt)
//   4) acc[t] += M strip @ UidT rows (K = 64 -> 2 WMMAs per e-tile).
// Writes out[...,384:512] = Uid * A_Q2D.
// ---------------------------------------------------------------------------
constexpr int KB = 64;                      // staged k-rows per block

__global__ void __launch_bounds__(128)
k_q2d(const __bf16* __restrict__ P, const __bf16* __restrict__ UidT,
      const float* __restrict__ Uid, float* __restrict__ out) {
  __shared__ __bf16 ldsP[KB * Lc];          // 256 KB staged P_k panel
  __shared__ __bf16 ldsM[4][16 * KB];       // per-wave M transpose (8 KB)

  const int wg  = blockIdx.x;               // B * (L/64)
  const int b   = wg / (Lc / 64);
  const int ib  = (wg % (Lc / 64)) * 64;
  const int tid = threadIdx.x;
  const int w   = tid >> 5;                 // wave 0..3
  const int l   = tid & 31, lm = l & 15, g = l >> 4;
  const int i0  = ib + w * 16;              // this wave's 16-row block

  const __bf16* P_b    = P    + (size_t)b * Lc * Lc;
  const __bf16* UidT_b = UidT + (size_t)b * Ec * Lc;

  v8f acc[8];
#pragma unroll
  for (int t = 0; t < 8; ++t) acc[t] = vzero8();

  const __bf16* arow = P_b + (size_t)(i0 + lm) * Lc;

  for (int k0 = 0; k0 < Lc; k0 += KB) {
    __syncthreads();  // all waves done reading ldsP from previous iteration
#if USE_TDM
    if (w == 0) {
      tdm_load_2d_bf16(P_b + (size_t)k0 * Lc,
                       (uint32_t)(uintptr_t)&ldsP[0], KB, Lc, Lc);
      __builtin_amdgcn_s_wait_tensorcnt(0);
    }
#else
    for (int c = tid; c < KB * Lc / 8; c += 128) {
      const int row  = c >> 8;              // Lc/8 = 256 chunks per row
      const int col8 = c & 255;
      *(v8bf*)&ldsP[(size_t)row * Lc + col8 * 8] =
          *(const v8bf*)&P_b[(size_t)(k0 + row) * Lc + col8 * 8];
    }
#endif
    __syncthreads();

    // M strip: 16 i-rows x 64 k-cols, contracting all j.
    v8f macc[4];
#pragma unroll
    for (int n = 0; n < 4; ++n) macc[n] = vzero8();
    for (int j0 = 0; j0 < Lc; j0 += 32) {
      const v16bf a = load_ab(arow + j0, g);          // global (L2-hot)
#pragma unroll
      for (int n = 0; n < 4; ++n) {
        const v16bf bb = load_ab(&ldsP[(size_t)(16 * n + lm) * Lc + j0], g);
        macc[n] = wmma_bf16(a, bb, macc[n]);
      }
    }

    // C layout: lane = k-col (16n+lm), VGPR r = i-row (r+8g). Store [i][k].
#pragma unroll
    for (int n = 0; n < 4; ++n)
#pragma unroll
      for (int r = 0; r < 8; ++r)
        ldsM[w][(r + 8 * g) * KB + 16 * n + lm] = (__bf16)macc[n][r];
    // ldsM is wave-private and DS ops are in-order within a wave: no barrier.
    const v16bf aM0 = load_ab(&ldsM[w][lm * KB], g);
    const v16bf aM1 = load_ab(&ldsM[w][lm * KB + 32], g);

#pragma unroll
    for (int t = 0; t < 8; ++t) {
      const __bf16* brow = UidT_b + (size_t)(t * 16 + lm) * Lc + k0;
      acc[t] = wmma_bf16(aM0, load_ab(brow, g), acc[t]);
      acc[t] = wmma_bf16(aM1, load_ab(brow + 32, g), acc[t]);
    }
  }

  const float* Uid_b = Uid + (size_t)b * Lc * Ec;
  float* out_b = out + (size_t)b * Lc * (4 * Ec);
#pragma unroll
  for (int t = 0; t < 8; ++t) {
#pragma unroll
    for (int r = 0; r < 8; ++r) {
      const int ir = i0 + r + 8 * g;
      const int e  = t * 16 + lm;
      const float u = Uid_b[(size_t)ir * Ec + e];
      out_b[(size_t)ir * (4 * Ec) + 3 * Ec + e] = u * acc[t][r];
    }
  }
}

// ---------------------------------------------------------------------------
// Host launcher. Workspace layout (~84.1 MB):
//   sid f32[B*L] | sq f32[B*L] | Uqb,Uwb,UqT,UidT bf16[B*L*E] | P bf16[B*L*L]
// ---------------------------------------------------------------------------
extern "C" void kernel_launch(void* const* d_in, const int* in_sizes, int n_in,
                              void* d_out, int out_size, void* d_ws, size_t ws_size,
                              hipStream_t stream) {
  const float* Uq   = (const float*)d_in[0];
  const float* Uid  = (const float*)d_in[1];
  const float* mask = (const float*)d_in[2];
  const float* Wcw  = (const float*)d_in[3];
  const float* Wcb  = (const float*)d_in[4];
  float* out = (float*)d_out;

  char* ws = (char*)d_ws;
  float* sid = (float*)ws;  ws += (size_t)Bc * Lc * sizeof(float);
  float* sq  = (float*)ws;  ws += (size_t)Bc * Lc * sizeof(float);
  __bf16* Uqb  = (__bf16*)ws;  ws += (size_t)Bc * Lc * Ec * sizeof(__bf16);
  __bf16* Uwb  = (__bf16*)ws;  ws += (size_t)Bc * Lc * Ec * sizeof(__bf16);
  __bf16* UqT  = (__bf16*)ws;  ws += (size_t)Bc * Lc * Ec * sizeof(__bf16);
  __bf16* UidT = (__bf16*)ws;  ws += (size_t)Bc * Lc * Ec * sizeof(__bf16);
  __bf16* P    = (__bf16*)ws;  // Bc*Lc*Lc elements

  k_prep<<<(Bc * Lc * Ec) / 256, 256, 0, stream>>>(Uq, Uid, Wcw, Uqb, Uwb, UqT, UidT);
  k_rowdots<<<(Bc * Lc) / 8, 256, 0, stream>>>(Uq, Uid, Wcw, sid, sq);
  k_scores_softmax<<<Bc * (Lc / 16), 32, 0, stream>>>(Uwb, Uqb, sid, sq, mask, Wcb, P);
  k_d2q<<<Bc * (Lc / 16), 32, 0, stream>>>(P, UqT, Uid, out);
  k_q2d<<<Bc * (Lc / 64), 128, 0, stream>>>(P, UidT, Uid, out);
}
